// PointNetSetAbstraction_4011499454875
// MI455X (gfx1250) — matrix operand order repaired and checked by reading, not compiled, not run on hardware
//
#include <hip/hip_runtime.h>
#include <hip/hip_bf16.h>

typedef __attribute__((ext_vector_type(16))) _Float16 v16h;
typedef __attribute__((ext_vector_type(8)))  _Float16 v8h;
typedef __attribute__((ext_vector_type(8)))  float    v8f;

#define B_ 16
#define N_ 4096
#define S_ 1024
#define K_ 32
#define M_ (B_ * S_ * K_)        // 524288 rows for the GEMMs
#define RAD2 0.04f
#define STAT_ROWS 4096           // rows per stats block
#define STAT_BLOCKS (M_ / STAT_ROWS)  // 128

// ---------------------------------------------------------------------------
// 1) Farthest-point sampling: one block per batch, 1024 sequential steps.
// ---------------------------------------------------------------------------
__global__ __launch_bounds__(256) void fps_kernel(const float* __restrict__ xyz,
                                                  int* __restrict__ fps_idx,
                                                  float* __restrict__ new_xyz) {
  __shared__ float lx[N_], ly[N_], lz[N_];
  __shared__ float rv[256];
  __shared__ int   ri[256];
  __shared__ int   s_far;
  const int b = blockIdx.x, tid = threadIdx.x;
  const float* xb = xyz + (size_t)b * N_ * 3;
  for (int i = tid; i < N_; i += 256) {
    lx[i] = xb[i * 3 + 0];
    ly[i] = xb[i * 3 + 1];
    lz[i] = xb[i * 3 + 2];
  }
  float dist[16];
#pragma unroll
  for (int j = 0; j < 16; ++j) dist[j] = 1e10f;
  int far = 0;
  __syncthreads();
  for (int it = 0; it < S_; ++it) {
    if (tid == 0) {
      fps_idx[b * S_ + it] = far;                 // emit carry BEFORE update (matches scan)
      new_xyz[(b * S_ + it) * 3 + 0] = lx[far];
      new_xyz[(b * S_ + it) * 3 + 1] = ly[far];
      new_xyz[(b * S_ + it) * 3 + 2] = lz[far];
    }
    const float cx = lx[far], cy = ly[far], cz = lz[far];
    float bv = -1.0f; int bi = 0;
#pragma unroll
    for (int j = 0; j < 16; ++j) {
      const int p = j * 256 + tid;
      const float dx = lx[p] - cx, dy = ly[p] - cy, dz = lz[p] - cz;
      const float d = dx * dx + dy * dy + dz * dz;
      if (d < dist[j]) dist[j] = d;
      const float dd = dist[j];
      if (dd > bv) { bv = dd; bi = p; }           // strict > keeps lowest index
    }
    rv[tid] = bv; ri[tid] = bi;
    __syncthreads();
    for (int off = 128; off; off >>= 1) {
      if (tid < off) {
        const float v2 = rv[tid + off]; const int i2 = ri[tid + off];
        if (v2 > rv[tid] || (v2 == rv[tid] && i2 < ri[tid])) { rv[tid] = v2; ri[tid] = i2; }
      }
      __syncthreads();
    }
    if (tid == 0) s_far = ri[0];
    __syncthreads();
    far = s_far;
    __syncthreads();
  }
}

// ---------------------------------------------------------------------------
// 2) Ball query: one wave per centroid; ballot-based ordered append.
// ---------------------------------------------------------------------------
__global__ __launch_bounds__(256) void ballquery_kernel(const float* __restrict__ xyz,
                                                        const float* __restrict__ new_xyz,
                                                        int* __restrict__ gidx) {
  const int w = blockIdx.x * 8 + (threadIdx.x >> 5);
  const int lane = threadIdx.x & 31;
  const int b = w >> 10;
  const float* xb = xyz + (size_t)b * N_ * 3;
  const float cx = new_xyz[w * 3 + 0];
  const float cy = new_xyz[w * 3 + 1];
  const float cz = new_xyz[w * 3 + 2];
  int cnt = 0, first = 0;
  for (int n0 = 0; n0 < N_ && cnt < K_; n0 += 32) {
    const int p = n0 + lane;
    const float dx = xb[p * 3 + 0] - cx;
    const float dy = xb[p * 3 + 1] - cy;
    const float dz = xb[p * 3 + 2] - cz;
    const bool in = (dx * dx + dy * dy + dz * dz) <= RAD2;
    const unsigned mask = __builtin_amdgcn_ballot_w32(in);
    if (cnt == 0 && mask) first = n0 + __builtin_ctz(mask);
    const int pos = cnt + __builtin_popcount(mask & ((1u << lane) - 1u));
    if (in && pos < K_) gidx[(size_t)w * K_ + pos] = p;
    cnt += __builtin_popcount(mask);
  }
  if (cnt < K_) {
    const int t = cnt + lane;
    if (t < K_) gidx[(size_t)w * K_ + t] = first;
  }
}

// ---------------------------------------------------------------------------
// 3) Weight prep: f32 (Cout x Cin) -> fragment-ready swizzled f16 buffer.
//    Layout: [(ks_idx*NT + tile_n)*32 + lane]*16 + h, so a wave's B fragment
//    for one k-step is a single contiguous 32-byte per-lane load.
//    h -> K mapping per ISA B 32x16 layout: K = ks + (h&7) + 8*(lane>>4) + 16*(h>>3).
// ---------------------------------------------------------------------------
__global__ void wconv_kernel(const float* __restrict__ w, _Float16* __restrict__ wf,
                             int CIN, int CIN_PAD, int COUT) {
  const int t = blockIdx.x * 256 + threadIdx.x;
  if (t >= CIN_PAD * COUT) return;
  const int NT = COUT / 16;
  const int h = t & 15;
  const int chunk = t >> 4;
  const int lane = chunk & 31;
  const int kt = chunk >> 5;
  const int tile_n = kt % NT;
  const int ks_idx = kt / NT;
  const int Kg = ks_idx * 32 + (h & 7) + ((lane >> 4) << 3) + ((h >> 3) << 4);
  const int o = tile_n * 16 + (lane & 15);
  wf[t] = (Kg < CIN) ? (_Float16)w[o * CIN + Kg] : (_Float16)0.f;
}

// ---------------------------------------------------------------------------
// 4) WMMA GEMM, LDS-staged A.
//    Block = 256 threads = 8 waves; computes (8/NT) M-tiles x COUT.
//    MODE 0: A gathered from xyz/points via group_idx (Cin 67 -> 96 pad)
//    MODE 1: A = prev f16 activations with fused BN (scale/shift) + ReLU
// ---------------------------------------------------------------------------
template <int MODE, int CIN_PAD, int COUT>
__global__ __launch_bounds__(256) void gemm_wmma_kernel(
    const _Float16* __restrict__ Ain, const float* __restrict__ scale,
    const float* __restrict__ shift, const float* __restrict__ xyz,
    const float* __restrict__ newxyz, const float* __restrict__ points,
    const int* __restrict__ gidx, const _Float16* __restrict__ Bw,
    _Float16* __restrict__ Yout) {
  constexpr int NT = COUT / 16;       // n-tiles per block
  constexpr int MT = 8 / NT;          // m-tiles per block
  constexpr int ROWS = 16 * MT;       // A rows staged per block
  __shared__ __align__(16) _Float16 As[ROWS][CIN_PAD];
  __shared__ int gi_l[ROWS];

  const int tid = threadIdx.x;
  const int lane = tid & 31;
  const int wv = tid >> 5;
  const int row0 = blockIdx.x * ROWS;

  // ---- Stage A into LDS (once; whole K range) ----
  if (MODE == 0) {
    if (tid < ROWS) gi_l[tid] = gidx[row0 + tid];
    __syncthreads();
    constexpr int CHUNKS = ROWS * CIN_PAD / 8;   // 8 halves per chunk
    for (int ch = tid; ch < CHUNKS; ch += 256) {
      const int r = ch / (CIN_PAD / 8);
      const int c0 = (ch % (CIN_PAD / 8)) * 8;
      const int row = row0 + r;
      const int b = row >> 15;                    // 32768 rows per batch
      const int s = (row & 32767) >> 5;
      const int gi = gi_l[r];
      const float* px = xyz + ((size_t)b * N_ + gi) * 3;
      const float* pc = newxyz + ((size_t)b * S_ + s) * 3;
      const float* pp = points + ((size_t)b * N_ + gi) * 64;
      v8h tmp;
#pragma unroll
      for (int i = 0; i < 8; ++i) {
        const int c = c0 + i;
        float v;
        if (c < 3)       v = px[c] - pc[c];
        else if (c < 67) v = pp[c - 3];
        else             v = 0.f;
        tmp[i] = (_Float16)v;
      }
      *(v8h*)&As[r][c0] = tmp;
    }
  } else {
    constexpr int CHUNKS = ROWS * CIN_PAD / 8;
    for (int ch = tid; ch < CHUNKS; ch += 256) {
      const int r = ch / (CIN_PAD / 8);
      const int c0 = (ch % (CIN_PAD / 8)) * 8;
      const int row = row0 + r;
      const v8h vin = *(const v8h*)(Ain + (size_t)row * CIN_PAD + c0);
      v8h tmp;
#pragma unroll
      for (int i = 0; i < 8; ++i) {
        float v = (float)vin[i];
        v = v * scale[c0 + i] + shift[c0 + i];
        tmp[i] = (_Float16)(v > 0.f ? v : 0.f);   // fused BN + ReLU
      }
      *(v8h*)&As[r][c0] = tmp;
    }
  }
  __syncthreads();

  // ---- Per-wave 16x16 tile ----
  const int tile_n = wv % NT;
  const int tile_ml = wv / NT;
  const int half = lane >> 4;
  const int l16 = lane & 15;
  const int arow = tile_ml * 16 + l16;
  const int col = tile_n * 16 + l16;

  v8f acc = {};
#pragma unroll
  for (int ks = 0; ks < CIN_PAD; ks += 32) {
    // A fragment: two contiguous 8-half runs from LDS (ISA A 16x32 layout)
    const v8h alo = *(const v8h*)&As[arow][ks + (half << 3)];
    const v8h ahi = *(const v8h*)&As[arow][ks + 16 + (half << 3)];
    const v16h a = __builtin_shufflevector(alo, ahi, 0, 1, 2, 3, 4, 5, 6, 7,
                                           8, 9, 10, 11, 12, 13, 14, 15);
    // B fragment: single contiguous per-lane load from swizzled weights
    const v16h bf = *(const v16h*)(Bw + (((size_t)(ks >> 5) * NT + tile_n) * 32 + lane) * 16);
    acc = __builtin_amdgcn_wmma_f32_16x16x32_f16(false, a, false, bf,
                                                 (short)0, acc, false, false);
  }
  // D: VGPR j holds row j + 8*half, column = lane&15
#pragma unroll
  for (int j = 0; j < 8; ++j) {
    const int rm = row0 + tile_ml * 16 + j + (half << 3);
    Yout[(size_t)rm * COUT + col] = (_Float16)acc[j];
  }
}

// ---------------------------------------------------------------------------
// 5) BN statistics: per-block partial sums (deterministic, no float atomics).
// ---------------------------------------------------------------------------
template <int COUT>
__global__ __launch_bounds__(256) void stats_kernel(const _Float16* __restrict__ Y,
                                                    float* __restrict__ psum,
                                                    float* __restrict__ psqs) {
  const int TPR = 256 / COUT;
  const int c = threadIdx.x % COUT;
  const int ro = threadIdx.x / COUT;
  const int r0 = blockIdx.x * STAT_ROWS;
  float s1 = 0.f, s2 = 0.f;
  for (int r = r0 + ro; r < r0 + STAT_ROWS; r += TPR) {
    const float v = (float)Y[(size_t)r * COUT + c];
    s1 += v; s2 += v * v;
  }
  __shared__ float a1[256], a2[256];
  a1[threadIdx.x] = s1; a2[threadIdx.x] = s2;
  __syncthreads();
  if (threadIdx.x < COUT) {
    for (int i = 1; i < TPR; ++i) { s1 += a1[c + i * COUT]; s2 += a2[c + i * COUT]; }
    psum[blockIdx.x * COUT + c] = s1;
    psqs[blockIdx.x * COUT + c] = s2;
  }
}

template <int COUT>
__global__ void finalize_bn_kernel(const float* __restrict__ psum,
                                   const float* __restrict__ psqs,
                                   const float* __restrict__ g,
                                   const float* __restrict__ be,
                                   float* __restrict__ scale,
                                   float* __restrict__ shift) {
  const int c = threadIdx.x;
  if (c >= COUT) return;
  float s1 = 0.f, s2 = 0.f;
  for (int i = 0; i < STAT_BLOCKS; ++i) { s1 += psum[i * COUT + c]; s2 += psqs[i * COUT + c]; }
  const float invM = 1.0f / (float)M_;
  const float mean = s1 * invM;
  const float var = s2 * invM - mean * mean;
  const float rstd = rsqrtf(var + 1e-5f);
  const float sc = g[c] * rstd;
  scale[c] = sc;
  shift[c] = be[c] - mean * sc;
}

// ---------------------------------------------------------------------------
// 6) Fused BN+ReLU + max over K=32 -> new_points (B,S,128).
// ---------------------------------------------------------------------------
__global__ __launch_bounds__(256) void maxpool_kernel(const _Float16* __restrict__ Y2,
                                                      const float* __restrict__ scale,
                                                      const float* __restrict__ shift,
                                                      float* __restrict__ out) {
  const int t = blockIdx.x * 256 + threadIdx.x;   // over B*S*128
  const int c = t & 127;
  const int bs = t >> 7;
  const size_t base = (size_t)bs * K_ * 128 + c;
  const float sc = scale[c], sh = shift[c];
  float m = -1e30f;
#pragma unroll 4
  for (int k = 0; k < K_; ++k) {
    const float v = (float)Y2[base + (size_t)k * 128] * sc + sh;
    if (v > m) m = v;
  }
  out[t] = m > 0.f ? m : 0.f;   // relu(max) == max(relu)
}

// ---------------------------------------------------------------------------
// Launch
// ---------------------------------------------------------------------------
extern "C" void kernel_launch(void* const* d_in, const int* in_sizes, int n_in,
                              void* d_out, int out_size, void* d_ws, size_t ws_size,
                              hipStream_t stream) {
  const float* xyz    = (const float*)d_in[0];
  const float* points = (const float*)d_in[1];
  const float* w0  = (const float*)d_in[2];
  const float* g0  = (const float*)d_in[4];
  const float* be0 = (const float*)d_in[5];
  const float* w1  = (const float*)d_in[6];
  const float* g1  = (const float*)d_in[8];
  const float* be1 = (const float*)d_in[9];
  const float* w2  = (const float*)d_in[10];
  const float* g2  = (const float*)d_in[12];
  const float* be2 = (const float*)d_in[13];

  float* new_xyz    = (float*)d_out;                 // (16,1024,3)
  float* new_points = (float*)d_out + B_ * S_ * 3;   // (16,1024,128)

  char* ws = (char*)d_ws;
  size_t off = 0;
  auto alloc = [&](size_t bytes) { void* p = ws + off; off = (off + bytes + 255) & ~(size_t)255; return p; };
  int*       fps_idx = (int*)alloc(B_ * S_ * sizeof(int));
  int*       gidx    = (int*)alloc((size_t)M_ * sizeof(int));
  _Float16*  wf0     = (_Float16*)alloc((size_t)96 * 64 * 2);
  _Float16*  wf1     = (_Float16*)alloc((size_t)64 * 64 * 2);
  _Float16*  wf2     = (_Float16*)alloc((size_t)64 * 128 * 2);
  float*     psum    = (float*)alloc((size_t)STAT_BLOCKS * 128 * sizeof(float));
  float*     psqs    = (float*)alloc((size_t)STAT_BLOCKS * 128 * sizeof(float));
  float*     sc0 = (float*)alloc(64 * sizeof(float));
  float*     sh0 = (float*)alloc(64 * sizeof(float));
  float*     sc1 = (float*)alloc(64 * sizeof(float));
  float*     sh1 = (float*)alloc(64 * sizeof(float));
  float*     sc2 = (float*)alloc(128 * sizeof(float));
  float*     sh2 = (float*)alloc(128 * sizeof(float));
  _Float16*  y0 = (_Float16*)alloc((size_t)M_ * 64 * 2);
  _Float16*  y1 = (_Float16*)alloc((size_t)M_ * 64 * 2);
  _Float16*  y2 = (_Float16*)alloc((size_t)M_ * 128 * 2);
  (void)ws_size; (void)in_sizes; (void)n_in; (void)out_size; (void)fps_idx;

  // 1) FPS (writes new_xyz output directly)
  fps_kernel<<<B_, 256, 0, stream>>>(xyz, fps_idx, new_xyz);
  // 2) Ball query
  ballquery_kernel<<<(B_ * S_) / 8, 256, 0, stream>>>(xyz, new_xyz, gidx);
  // 3) Weight conversion (f32 -> fragment-ready swizzled f16)
  wconv_kernel<<<(96 * 64 + 255) / 256, 256, 0, stream>>>(w0, wf0, 67, 96, 64);
  wconv_kernel<<<(64 * 64 + 255) / 256, 256, 0, stream>>>(w1, wf1, 64, 64, 64);
  wconv_kernel<<<(64 * 128 + 255) / 256, 256, 0, stream>>>(w2, wf2, 64, 64, 128);

  const int TM = M_ / 16;  // 32768 M-tiles
  // 4) Layer 0: gather + GEMM (Cin 67 padded to 96); 2 M-tiles per block
  gemm_wmma_kernel<0, 96, 64><<<TM / 2, 256, 0, stream>>>(
      nullptr, nullptr, nullptr, xyz, new_xyz, points, gidx, wf0, y0);
  stats_kernel<64><<<STAT_BLOCKS, 256, 0, stream>>>(y0, psum, psqs);
  finalize_bn_kernel<64><<<1, 64, 0, stream>>>(psum, psqs, g0, be0, sc0, sh0);
  // 5) Layer 1: fused BN0+ReLU on load
  gemm_wmma_kernel<1, 64, 64><<<TM / 2, 256, 0, stream>>>(
      y0, sc0, sh0, nullptr, nullptr, nullptr, nullptr, wf1, y1);
  stats_kernel<64><<<STAT_BLOCKS, 256, 0, stream>>>(y1, psum, psqs);
  finalize_bn_kernel<64><<<1, 64, 0, stream>>>(psum, psqs, g1, be1, sc1, sh1);
  // 6) Layer 2: fused BN1+ReLU on load; 1 M-tile per block
  gemm_wmma_kernel<1, 64, 128><<<TM, 256, 0, stream>>>(
      y1, sc1, sh1, nullptr, nullptr, nullptr, nullptr, wf2, y2);
  stats_kernel<128><<<STAT_BLOCKS, 256, 0, stream>>>(y2, psum, psqs);
  finalize_bn_kernel<128><<<1, 128, 0, stream>>>(psum, psqs, g2, be2, sc2, sh2);
  // 7) Fused BN2+ReLU + max over K
  maxpool_kernel<<<(B_ * S_ * 128) / 256, 256, 0, stream>>>(y2, sc2, sh2, new_points);
}
